// Pseudo_Label_Loss_43989055045817
// MI455X (gfx1250) — matrix operand order repaired and checked by reading, not compiled, not run on hardware
//
#include <hip/hip_runtime.h>
#include <hip/hip_bf16.h>
#include <math.h>

typedef __attribute__((ext_vector_type(16))) _Float16 v16h;
typedef __attribute__((ext_vector_type(8)))  _Float16 v8h;
typedef __attribute__((ext_vector_type(8)))  float    v8f;

#define N_ROWS 8192
#define DIMS   256
#define NCLS   100
#define NTILE  (N_ROWS / 16)   // 512
#define TAU_INV 2.0f
#define EPS 1e-8

// ---------------------------------------------------------------------------
// 0) zero the three double accumulators (neg_sum, pos_sum, pos_cnt)
// ---------------------------------------------------------------------------
__global__ void init_acc_kernel(double* __restrict__ acc) {
    acc[0] = 0.0;
    acc[1] = 0.0;
    acc[2] = 0.0;
}

// ---------------------------------------------------------------------------
// 1) lab[r] = argmax_c label[r, c]   (first-max tiebreak like jnp.argmax)
// ---------------------------------------------------------------------------
__global__ void argmax_kernel(const float* __restrict__ label, int* __restrict__ lab) {
    int r = blockIdx.x * blockDim.x + threadIdx.x;
    if (r >= N_ROWS) return;
    const float* row = label + (size_t)r * NCLS;
    float best = row[0];
    int   bi   = 0;
    for (int c = 1; c < NCLS; ++c) {
        float v = row[c];
        if (v > best) { best = v; bi = c; }
    }
    lab[r] = bi;
}

// ---------------------------------------------------------------------------
// 2) fnh[r,:] = (f16) feature[r,:] / max(||feature[r,:]||, 1e-8)
//    one wave32 per row; 8 elements per lane; shuffle reduction
// ---------------------------------------------------------------------------
__global__ void normalize_kernel(const float* __restrict__ feat,
                                 _Float16* __restrict__ fnh) {
    int wave = (blockIdx.x * blockDim.x + threadIdx.x) >> 5;
    int lane = threadIdx.x & 31;
    if (wave >= N_ROWS) return;
    const float* row = feat + (size_t)wave * DIMS;
    float vals[8];
    float ss = 0.0f;
#pragma unroll
    for (int u = 0; u < 8; ++u) {
        float f = row[lane + 32 * u];
        vals[u] = f;
        ss += f * f;
    }
#pragma unroll
    for (int m = 16; m > 0; m >>= 1) ss += __shfl_xor(ss, m, 32);
    float nrm = fmaxf(sqrtf(ss), 1e-8f);
    float r = 1.0f / nrm;
    _Float16* out = fnh + (size_t)wave * DIMS;
#pragma unroll
    for (int u = 0; u < 8; ++u) out[lane + 32 * u] = (_Float16)(vals[u] * r);
}

// ---------------------------------------------------------------------------
// 3) main kernel: upper-triangular tiled WMMA GEMM + masked reductions.
//    Grid = 256 blocks x 256 threads (8 waves). Block b owns TWO row tiles:
//    it = b and it = 511 - b  -> exactly 513 tiles per block (perfect balance).
//    Wave w handles column tiles jt = it + w, it + w + 8, ...
//    A-tile (16 rows x K=256 f16, 64 VGPRs) loaded once per row tile, reused.
//    B tile staged fully in registers before the WMMA chain so loads clause
//    up and overlap with matrix work. Off-diagonal tiles weighted x2.
// ---------------------------------------------------------------------------
__global__ __launch_bounds__(256)
void simloss_kernel(const _Float16* __restrict__ fnh,
                    const int* __restrict__ lab,
                    double* __restrict__ acc) {
    const int lane = threadIdx.x & 31;
    const int w    = threadIdx.x >> 5;     // wave id in block (uniform per wave)

    const int mOff = (lane >> 4) << 3;     // C/D: VGPR v holds M = v + mOff
    const int nIdx = lane & 15;            // C/D + B: column index
    const int kbA  = (lane < 16) ? 0 : 16; // A K-split: contiguous, matches B
    const int khB  = (lane < 16) ? 0 : 16; // B K-split: contiguous, matches A

    float negAcc = 0.0f, posAcc = 0.0f, cntAcc = 0.0f;

    for (int pass = 0; pass < 2; ++pass) {
        const int it = pass ? (NTILE - 1 - (int)blockIdx.x) : (int)blockIdx.x;

        // ---- load full-K A tile: lane holds row (it*16 + lane&15)
        const _Float16* aPtr = fnh + (size_t)(it * 16 + nIdx) * DIMS + kbA;
        v16h aReg[8];
#pragma unroll
        for (int s = 0; s < 8; ++s) {
            v8h lo = *(const v8h*)(aPtr + 32 * s);        // K = 32s+kb .. +7
            v8h hi = *(const v8h*)(aPtr + 32 * s + 8);    // K = 32s+kb+8 .. +15
#pragma unroll
            for (int x = 0; x < 8; ++x) { aReg[s][x] = lo[x]; aReg[s][8 + x] = hi[x]; }
        }

        // labels for this lane's 8 output rows
        int labi[8];
#pragma unroll
        for (int v = 0; v < 8; ++v) labi[v] = lab[it * 16 + v + mOff];

        for (int jt = it + w; jt < NTILE; jt += 8) {      // uniform per wave
            const int jRow = jt * 16 + nIdx;
            const _Float16* bPtr = fnh + (size_t)jRow * DIMS + khB;

            // stage whole B tile first: 16 x b128 loads in one clause,
            // partial loadcnt waits then overlap with the WMMA chain
            v16h bReg[8];
#pragma unroll
            for (int s = 0; s < 8; ++s)
                bReg[s] = *(const v16h*)(bPtr + 32 * s);

            // prefetch next j-tile's B rows (uniform condition)
            if (jt + 8 < NTILE)
                __builtin_prefetch(bPtr + 128 * DIMS, 0, 0);

            v8f c = {};
#pragma unroll
            for (int s = 0; s < 8; ++s)
                c = __builtin_amdgcn_wmma_f32_16x16x32_f16(
                        false, aReg[s], false, bReg[s], (short)0, c, false, false);

            const int   labj = lab[jRow];
            const float wgt  = (jt == it) ? 1.0f : 2.0f;  // symmetry doubling
#pragma unroll
            for (int v = 0; v < 8; ++v) {
                int   i    = it * 16 + v + mOff;
                float sim  = c[v] * TAU_INV;
                bool  off  = (i != jRow);
                bool  same = (labi[v] == labj);
                if (off && same)  { posAcc += wgt * sim; cntAcc += wgt; }
                if (off && !same) { negAcc += wgt * __expf(sim); }
            }
        }
    }

    // wave32 reduction, then one atomic triple per wave
#pragma unroll
    for (int m = 16; m > 0; m >>= 1) {
        negAcc += __shfl_xor(negAcc, m, 32);
        posAcc += __shfl_xor(posAcc, m, 32);
        cntAcc += __shfl_xor(cntAcc, m, 32);
    }
    if (lane == 0) {
        atomicAdd(&acc[0], (double)negAcc);
        atomicAdd(&acc[1], (double)posAcc);
        atomicAdd(&acc[2], (double)cntAcc);
    }
}

// ---------------------------------------------------------------------------
// 4) loss = log(neg_sum + eps) - pos_sum / pos_cnt
// ---------------------------------------------------------------------------
__global__ void finalize_kernel(const double* __restrict__ acc,
                                float* __restrict__ out) {
    double neg = acc[0], pos = acc[1], cnt = acc[2];
    out[0] = (float)(log(neg + EPS) - pos / cnt);
}

// ---------------------------------------------------------------------------
extern "C" void kernel_launch(void* const* d_in, const int* in_sizes, int n_in,
                              void* d_out, int out_size, void* d_ws, size_t ws_size,
                              hipStream_t stream) {
    (void)in_sizes; (void)n_in; (void)out_size; (void)ws_size;
    const float* feature = (const float*)d_in[0];   // [8192, 256]
    const float* label   = (const float*)d_in[1];   // [8192, 100]
    float*       out     = (float*)d_out;

    // workspace layout: [0..63] doubles acc, then lab (32KB), then fnh (4MB)
    double*   acc = (double*)d_ws;
    int*      lab = (int*)((char*)d_ws + 64);
    _Float16* fnh = (_Float16*)((char*)d_ws + 64 + N_ROWS * sizeof(int));

    init_acc_kernel<<<1, 1, 0, stream>>>(acc);
    argmax_kernel<<<(N_ROWS + 255) / 256, 256, 0, stream>>>(label, lab);
    normalize_kernel<<<(N_ROWS * 32) / 256, 256, 0, stream>>>(feature, fnh);
    simloss_kernel<<<NTILE / 2, 256, 0, stream>>>(fnh, lab, acc);
    finalize_kernel<<<1, 1, 0, stream>>>(acc, out);
}
